// HC_UTAL_72928544686592
// MI455X (gfx1250) — matrix operand order repaired and checked by reading, not compiled, not run on hardware
//
#include <hip/hip_runtime.h>

// ---------------- problem constants ----------------
constexpr int B_   = 16;
constexpr int T_   = 750;
constexpr int D_   = 2048;   // embed dim / conv out channels
constexpr int CIN  = 1024;   // per-stream input channels
constexpr int NCLS = 20;
constexpr int KNB  = 150;    // T // 5
constexpr int KBD  = 37;     // T // 20
constexpr int TT   = 47;     // ceil(750/16) t-tiles

// ---------------- d_out layout (floats, concat in return order) ------------
constexpr size_t OFF_VS  = 0;                                  // [16,20]
constexpr size_t OFF_EA  = OFF_VS  + (size_t)B_ * NCLS;        // [16,150,2048]
constexpr size_t OFF_EB  = OFF_EA  + (size_t)B_ * KNB * D_;
constexpr size_t OFF_HA  = OFF_EB  + (size_t)B_ * KNB * D_;    // [16,37,2048]
constexpr size_t OFF_HB  = OFF_HA  + (size_t)B_ * KBD * D_;
constexpr size_t OFF_IDX = OFF_HB  + (size_t)B_ * KBD * D_;    // [16,150] (float)
constexpr size_t OFF_ACT = OFF_IDX + (size_t)B_ * KNB;         // [16,750]
constexpr size_t OFF_CAS = OFF_ACT + (size_t)B_ * T_;          // [16,750,20]
constexpr size_t OFF_EMB = OFF_CAS + (size_t)B_ * T_ * NCLS;   // [16,750,2048]

// ---------------- workspace layout (bytes) ----------------
constexpr size_t WS_WPT    = 0;                                   // bf16 [2][2048][3][1024]
constexpr size_t WS_WCLS   = WS_WPT  + (size_t)2*D_*3*CIN*2;      // bf16 [32][2048]
constexpr size_t WS_IDXEA  = WS_WCLS + (size_t)32*D_*2;           // int [16][150]
constexpr size_t WS_IDXEB  = WS_IDXEA + (size_t)B_*KNB*4;
constexpr size_t WS_IDXHA  = WS_IDXEB + (size_t)B_*KNB*4;
constexpr size_t WS_IDXHB  = WS_IDXHA + (size_t)B_*KBD*4;

typedef __attribute__((ext_vector_type(16))) __bf16 v16bf;
typedef __attribute__((ext_vector_type(8)))  float  v8f;

// ===================== weight packing =====================
// wpT[s][co][tap][ci] = bf16(w_s[co][ci][tap])
__global__ __launch_bounds__(256) void pack_w_kernel(
    const float* __restrict__ wr, const float* __restrict__ wf,
    __bf16* __restrict__ wpT)
{
    size_t i = (size_t)blockIdx.x * 256 + threadIdx.x;   // < 2*2048*3*1024
    int ci  = (int)(i & 1023);
    int r   = (int)(i >> 10);
    int tap = r % 3;
    int q   = r / 3;
    int co  = q & (D_ - 1);
    int s   = q >> 11;
    const float* w = s ? wf : wr;
    wpT[i] = (__bf16)w[((size_t)co * CIN + ci) * 3 + tap];
}

// wclsP[c][co] for c<20 else 0 (classes padded to 32)
__global__ __launch_bounds__(256) void pack_wcls_kernel(
    const float* __restrict__ wc, __bf16* __restrict__ wclsP)
{
    size_t i = (size_t)blockIdx.x * 256 + threadIdx.x;   // < 32*2048
    int c  = (int)(i >> 11);
    int co = (int)(i & (D_ - 1));
    float v = (c < NCLS) ? wc[(size_t)c * D_ + co] : 0.0f;
    wclsP[i] = (__bf16)v;
}

// build a 16-bit-A-layout fragment from 16 consecutive-K floats (one 16-float run)
__device__ __forceinline__ v16bf make_afrag(const float* p, float vscale)
{
    float4 q0 = *(const float4*)(p);
    float4 q1 = *(const float4*)(p + 4);
    float4 q2 = *(const float4*)(p + 8);
    float4 q3 = *(const float4*)(p + 12);
    v16bf a;
    a[0]  = (__bf16)(q0.x * vscale); a[1]  = (__bf16)(q0.y * vscale);
    a[2]  = (__bf16)(q0.z * vscale); a[3]  = (__bf16)(q0.w * vscale);
    a[4]  = (__bf16)(q1.x * vscale); a[5]  = (__bf16)(q1.y * vscale);
    a[6]  = (__bf16)(q1.z * vscale); a[7]  = (__bf16)(q1.w * vscale);
    a[8]  = (__bf16)(q2.x * vscale); a[9]  = (__bf16)(q2.y * vscale);
    a[10] = (__bf16)(q2.z * vscale); a[11] = (__bf16)(q2.w * vscale);
    a[12] = (__bf16)(q3.x * vscale); a[13] = (__bf16)(q3.y * vscale);
    a[14] = (__bf16)(q3.z * vscale); a[15] = (__bf16)(q3.w * vscale);
    return a;
}

// ===================== conv1d(k=3) dual-stream GEMM via WMMA =====================
// grid: (B, 47 t-tiles, 4 col-supertiles of 512); block 256 = 8 waves * 64 cols.
__global__ __launch_bounds__(256) void conv_wmma_kernel(
    const float* __restrict__ x,            // [B,T,2048]
    const __bf16* __restrict__ wpT,         // bf16 [2][2048][3][1024]
    const float* __restrict__ b_rgb, const float* __restrict__ b_flow,
    float* __restrict__ emb)                // [B,T,2048]
{
    const int b    = blockIdx.x;
    const int t0   = blockIdx.y * 16;
    const int wave = threadIdx.x >> 5;
    const int lane = threadIdx.x & 31;
    const int coW  = blockIdx.z * 512 + wave * 64;
    const int mrow = lane & 15;
    const int kh   = lane >> 4;

    v8f acc[2][4];
#pragma unroll
    for (int s = 0; s < 2; ++s)
#pragma unroll
        for (int j = 0; j < 4; ++j)
#pragma unroll
            for (int r = 0; r < 8; ++r) acc[s][j][r] = 0.0f;

    for (int s = 0; s < 2; ++s) {
        const float* xb = x + (size_t)b * T_ * D_ + s * CIN;
        for (int tap = 0; tap < 3; ++tap) {
            int row = t0 + mrow + tap - 1;
            bool valid = (row >= 0) && (row < T_);
            float vscale = valid ? 1.0f : 0.0f;
            const float* xr = xb + (size_t)(valid ? row : 0) * D_;
            // per-lane weight base for the 4 column subtiles
            const __bf16* wl0 = wpT + (((size_t)(s * D_ + coW + mrow)) * 3 + tap) * CIN + kh * 16;
            for (int kk = 0; kk < CIN; kk += 32) {
                v16bf af = make_afrag(xr + kk + kh * 16, vscale);
#pragma unroll
                for (int j = 0; j < 4; ++j) {
                    const __bf16* wp = wl0 + (size_t)j * 16 * 3 * CIN + kk;
                    v16bf bf = *(const v16bf*)wp;   // 32B contiguous
                    acc[s][j] = __builtin_amdgcn_wmma_f32_16x16x32_bf16(
                        false, af, false, bf, (short)0, acc[s][j], false, false);
                }
            }
        }
    }

    // epilogue: relu(rgb + b_rgb) + relu(flow + b_flow)
    const int n = lane & 15;
#pragma unroll
    for (int j = 0; j < 4; ++j) {
        int co = coW + j * 16 + n;
        float br = b_rgb[co], bl = b_flow[co];
#pragma unroll
        for (int r = 0; r < 8; ++r) {
            int t = t0 + r + 8 * kh;
            if (t < T_) {
                float v = fmaxf(acc[0][j][r] + br, 0.0f) + fmaxf(acc[1][j][r] + bl, 0.0f);
                emb[((size_t)b * T_ + t) * D_ + co] = v;
            }
        }
    }
}

// ===================== cas + actionness via WMMA =====================
// grid: (B, 6); block 256 = 8 waves, wave handles one 16-row t-tile, N=32 classes.
__global__ __launch_bounds__(256) void cas_wmma_kernel(
    const float* __restrict__ emb,
    const __bf16* __restrict__ wclsP,  // bf16 [32][2048]
    float* __restrict__ cas, float* __restrict__ act)
{
    const int b    = blockIdx.x;
    const int wave = threadIdx.x >> 5;
    const int lane = threadIdx.x & 31;
    const int tt   = blockIdx.y * 8 + wave;
    if (tt >= TT) return;
    const int t0   = tt * 16;
    const int mrow = lane & 15;
    const int kh   = lane >> 4;

    v8f acc0, acc1;
#pragma unroll
    for (int r = 0; r < 8; ++r) { acc0[r] = 0.0f; acc1[r] = 0.0f; }

    int arow = t0 + mrow;
    bool valid = arow < T_;
    float vscale = valid ? 1.0f : 0.0f;
    const float* xr = emb + ((size_t)b * T_ + (valid ? arow : 0)) * D_;
    const __bf16* w0 = wclsP + (size_t)mrow * D_ + kh * 16;
    const __bf16* w1 = wclsP + (size_t)(16 + mrow) * D_ + kh * 16;

    for (int kk = 0; kk < D_; kk += 32) {
        v16bf af = make_afrag(xr + kk + kh * 16, vscale);
        v16bf b0 = *(const v16bf*)(w0 + kk);
        v16bf b1 = *(const v16bf*)(w1 + kk);
        acc0 = __builtin_amdgcn_wmma_f32_16x16x32_bf16(false, af, false, b0, (short)0, acc0, false, false);
        acc1 = __builtin_amdgcn_wmma_f32_16x16x32_bf16(false, af, false, b1, (short)0, acc1, false, false);
    }

    const int n = lane & 15;
#pragma unroll
    for (int r = 0; r < 8; ++r) {
        int t = t0 + r + 8 * kh;
        float v0 = fmaxf(acc0[r], 0.0f);
        float v1 = fmaxf(acc1[r], 0.0f);       // classes 20..31 are exactly 0
        if (t < T_) {
            cas[((size_t)b * T_ + t) * NCLS + n] = v0;
            if (n < 4) cas[((size_t)b * T_ + t) * NCLS + 16 + n] = v1;
        }
        float s = v0 + v1;
        s += __shfl_xor(s, 1, 16);
        s += __shfl_xor(s, 2, 16);
        s += __shfl_xor(s, 4, 16);
        s += __shfl_xor(s, 8, 16);
        if (n == 0 && t < T_) act[(size_t)b * T_ + t] = s;
    }
}

// ===================== per-batch post-processing =====================
__device__ __forceinline__ void topk_rank(const float* s, int k, int* out, int tid)
{
    for (int i = tid; i < T_; i += 256) {
        float v = s[i];
        int r = 0;
        for (int j = 0; j < T_; ++j) {
            float u = s[j];
            r += (u > v) || ((u == v) && (j < i));
        }
        if (r < k) out[r] = i;
    }
}

__global__ __launch_bounds__(256) void post_kernel(
    const float* __restrict__ act, const float* __restrict__ cas,
    float* __restrict__ vs, float* __restrict__ idx_out,
    int* __restrict__ idxEA, int* __restrict__ idxEB,
    int* __restrict__ idxHA, int* __restrict__ idxHB)
{
    __shared__ float a[T_];
    __shared__ float sc[T_];
    __shared__ float bin[T_];
    __shared__ float red[256];
    __shared__ float s_med0, s_med1, s_sum;
    __shared__ float cls_mean[NCLS];

    const int b = blockIdx.x, tid = threadIdx.x;

    for (int i = tid; i < T_; i += 256) a[i] = act[(size_t)b * T_ + i];
    __syncthreads();

    // max (for reversed scores)
    float m = -3.4e38f;
    for (int i = tid; i < T_; i += 256) m = fmaxf(m, a[i]);
    red[tid] = m; __syncthreads();
    for (int s = 128; s > 0; s >>= 1) { if (tid < s) red[tid] = fmaxf(red[tid], red[tid + s]); __syncthreads(); }
    float amax = red[0];
    __syncthreads();

    // median = avg of sorted ranks 374, 375 (rank counting)
    for (int i = tid; i < T_; i += 256) {
        float v = a[i]; int less = 0, eq = 0;
        for (int j = 0; j < T_; ++j) { float u = a[j]; less += (u < v); eq += (u == v); }
        if (less <= 374 && 374 < less + eq) s_med0 = v;
        if (less <= 375 && 375 < less + eq) s_med1 = v;
    }
    __syncthreads();
    float med = 0.5f * (s_med0 + s_med1);

    // --- EA: top-150 of actionness (also the idx output) ---
    topk_rank(a, KNB, idxEA + b * KNB, tid);
    __syncthreads();
    for (int r = tid; r < KNB; r += 256) idx_out[(size_t)b * KNB + r] = (float)idxEA[b * KNB + r];

    // --- EB: top-150 of (max - actionness) ---
    for (int i = tid; i < T_; i += 256) sc[i] = amax - a[i];
    __syncthreads();
    topk_rank(sc, KNB, idxEB + b * KNB, tid);

    // binarize
    for (int i = tid; i < T_; i += 256) bin[i] = (a[i] > med) ? 1.0f : 0.0f;
    __syncthreads();

    // inner = erode3 - erode6 ; score = actionness * inner
    for (int i = tid; i < T_; i += 256) {
        float e3 = 1.0f, e6 = 1.0f;
        for (int j = i - 1; j <= i + 1; ++j) e3 = fminf(e3, (j >= 0 && j < T_) ? bin[j] : 0.0f);
        for (int j = i - 3; j <= i + 2; ++j) e6 = fminf(e6, (j >= 0 && j < T_) ? bin[j] : 0.0f);
        sc[i] = a[i] * (e3 - e6);
    }
    __syncthreads();
    topk_rank(sc, KBD, idxHA + b * KBD, tid);
    __syncthreads();

    // outer = dilate6 - dilate3 ; score = actionness * outer
    for (int i = tid; i < T_; i += 256) {
        float d6 = 0.0f, d3 = 0.0f;
        for (int j = i - 2; j <= i + 3; ++j) d6 = fmaxf(d6, (j >= 0 && j < T_) ? bin[j] : 0.0f);
        for (int j = i - 1; j <= i + 1; ++j) d3 = fmaxf(d3, (j >= 0 && j < T_) ? bin[j] : 0.0f);
        sc[i] = a[i] * (d6 - d3);
    }
    __syncthreads();
    topk_rank(sc, KBD, idxHB + b * KBD, tid);
    __syncthreads();

    // video-level: per-class mean of top-150 over time, then softmax
    for (int c = 0; c < NCLS; ++c) {
        for (int i = tid; i < T_; i += 256) sc[i] = cas[((size_t)b * T_ + i) * NCLS + c];
        if (tid == 0) s_sum = 0.0f;
        __syncthreads();
        for (int i = tid; i < T_; i += 256) {
            float v = sc[i]; int r = 0;
            for (int j = 0; j < T_; ++j) { float u = sc[j]; r += (u > v) || ((u == v) && (j < i)); }
            if (r < KNB) atomicAdd(&s_sum, v);
        }
        __syncthreads();
        if (tid == 0) cls_mean[c] = s_sum * (1.0f / KNB);
        __syncthreads();
    }
    if (tid == 0) {
        float mx = -3.4e38f;
        for (int c = 0; c < NCLS; ++c) mx = fmaxf(mx, cls_mean[c]);
        float e[NCLS], se = 0.0f;
        for (int c = 0; c < NCLS; ++c) { e[c] = __expf(cls_mean[c] - mx); se += e[c]; }
        float inv = 1.0f / se;
        for (int c = 0; c < NCLS; ++c) vs[(size_t)b * NCLS + c] = e[c] * inv;
    }
}

// ===================== gather selected embedding rows =====================
__global__ __launch_bounds__(256) void gather_kernel(
    const float* __restrict__ emb,
    const int* __restrict__ idxEA, const int* __restrict__ idxEB,
    const int* __restrict__ idxHA, const int* __restrict__ idxHB,
    float* __restrict__ outEA, float* __restrict__ outEB,
    float* __restrict__ outHA, float* __restrict__ outHB)
{
    int row = blockIdx.x;                        // 0 .. 5983
    const int nEA = B_ * KNB, nEB = B_ * KNB, nHA = B_ * KBD;
    int b, r, t; float* dst;
    if (row < nEA) {
        b = row / KNB; r = row % KNB; t = idxEA[row];
        dst = outEA + ((size_t)b * KNB + r) * D_;
    } else if (row < nEA + nEB) {
        row -= nEA; b = row / KNB; r = row % KNB; t = idxEB[row];
        dst = outEB + ((size_t)b * KNB + r) * D_;
    } else if (row < nEA + nEB + nHA) {
        row -= nEA + nEB; b = row / KBD; r = row % KBD; t = idxHA[row];
        dst = outHA + ((size_t)b * KBD + r) * D_;
    } else {
        row -= nEA + nEB + nHA; b = row / KBD; r = row % KBD; t = idxHB[row];
        dst = outHB + ((size_t)b * KBD + r) * D_;
    }
    const float* src = emb + ((size_t)b * T_ + t) * D_;
    for (int i = threadIdx.x; i < D_; i += 256) dst[i] = src[i];
}

// ===================== launch =====================
extern "C" void kernel_launch(void* const* d_in, const int* in_sizes, int n_in,
                              void* d_out, int out_size, void* d_ws, size_t ws_size,
                              hipStream_t stream)
{
    const float* x      = (const float*)d_in[0];
    const float* w_rgb  = (const float*)d_in[1];
    const float* b_rgb  = (const float*)d_in[2];
    const float* w_flow = (const float*)d_in[3];
    const float* b_flow = (const float*)d_in[4];
    const float* w_cls  = (const float*)d_in[5];
    float* out = (float*)d_out;

    __bf16* wpT   = (__bf16*)((char*)d_ws + WS_WPT);
    __bf16* wclsP = (__bf16*)((char*)d_ws + WS_WCLS);
    int* idxEA = (int*)((char*)d_ws + WS_IDXEA);
    int* idxEB = (int*)((char*)d_ws + WS_IDXEB);
    int* idxHA = (int*)((char*)d_ws + WS_IDXHA);
    int* idxHB = (int*)((char*)d_ws + WS_IDXHB);

    float* vs   = out + OFF_VS;
    float* ea   = out + OFF_EA;
    float* eb   = out + OFF_EB;
    float* ha   = out + OFF_HA;
    float* hb   = out + OFF_HB;
    float* idxf = out + OFF_IDX;
    float* act  = out + OFF_ACT;
    float* cas  = out + OFF_CAS;
    float* emb  = out + OFF_EMB;

    // 1. pack conv weights to bf16 [2][2048][3][1024]
    pack_w_kernel<<<(2 * D_ * 3 * CIN) / 256, 256, 0, stream>>>(w_rgb, w_flow, wpT);
    // 2. pack classifier weights to bf16 [32][2048] (padded)
    pack_wcls_kernel<<<(32 * D_) / 256, 256, 0, stream>>>(w_cls, wclsP);
    // 3. dual-stream conv GEMM -> embeddings
    conv_wmma_kernel<<<dim3(B_, TT, 4), 256, 0, stream>>>(x, wpT, b_rgb, b_flow, emb);
    // 4. cas + actionness
    cas_wmma_kernel<<<dim3(B_, 6), 256, 0, stream>>>(emb, wclsP, cas, act);
    // 5. per-batch median/morphology/top-k/video-scores
    post_kernel<<<B_, 256, 0, stream>>>(act, cas, vs, idxf, idxEA, idxEB, idxHA, idxHB);
    // 6. gather selected embedding rows
    gather_kernel<<<2 * B_ * KNB + 2 * B_ * KBD, 256, 0, stream>>>(
        emb, idxEA, idxEB, idxHA, idxHB, ea, eb, ha, hb);
}
